// CusparseDynamicLinear_72567767433792
// MI455X (gfx1250) — matrix operand order repaired and checked by reading, not compiled, not run on hardware
//
#include <hip/hip_runtime.h>

typedef __bf16 bf16_t;
typedef __attribute__((ext_vector_type(8)))  __bf16 v8bf;
typedef __attribute__((ext_vector_type(16))) __bf16 v16bf;
typedef __attribute__((ext_vector_type(8)))  float  v8f;

#define BM 128
#define BN 128
#define BK 32
#define LDSS 40          // padded LDS row stride in bf16 elems (80 B = 20 dwords)
#define NTHREADS 256

__device__ __forceinline__ v8bf pack8(float4 x, float4 y) {
    v8bf r;
    r[0] = (__bf16)x.x; r[1] = (__bf16)x.y; r[2] = (__bf16)x.z; r[3] = (__bf16)x.w;
    r[4] = (__bf16)y.x; r[5] = (__bf16)y.y; r[6] = (__bf16)y.z; r[7] = (__bf16)y.w;
    return r;
}

__device__ __forceinline__ float4 mul4(float4 a, float4 b) {
    return make_float4(a.x * b.x, a.y * b.y, a.z * b.z, a.w * b.w);
}

// A fragment (this code's contiguous k order, same as B): lane holds row M=lane&15,
// K = c0..c0+15 consecutive with c0 = (lane>>4)*16 -> one 32B run.
__device__ __forceinline__ v16bf load_fragA(const bf16_t* p) {
    v8bf lo = *reinterpret_cast<const v8bf*>(p);
    v8bf hi = *reinterpret_cast<const v8bf*>(p + 8);
    return __builtin_shufflevector(lo, hi, 0, 1, 2, 3, 4, 5, 6, 7,
                                   8, 9, 10, 11, 12, 13, 14, 15);
}

// B fragment (this code's contiguous k order, same as A): lane holds col N=lane&15,
// K = k0..k0+15 consecutive with k0 = (lane>>4)*16 -> one 32B run.
__device__ __forceinline__ v16bf load_fragB(const bf16_t* p) {
    v8bf lo = *reinterpret_cast<const v8bf*>(p);
    v8bf hi = *reinterpret_cast<const v8bf*>(p + 8);
    return __builtin_shufflevector(lo, hi, 0, 1, 2, 3, 4, 5, 6, 7,
                                   8, 9, 10, 11, 12, 13, 14, 15);
}

__global__ __launch_bounds__(NTHREADS)
void masked_linear_wmma_bf16(const float* __restrict__ data,
                             const float* __restrict__ weight,
                             const float* __restrict__ wmask,
                             const float* __restrict__ bias,
                             float* __restrict__ out,
                             int M, int N, int K) {
    __shared__ __align__(16) bf16_t As[2][BM * LDSS];
    __shared__ __align__(16) bf16_t Bs[2][BN * LDSS];

    const int tid   = threadIdx.x;
    const int lane  = tid & 31;
    const int wave  = tid >> 5;
    const int waveM = wave & 3;   // 4 waves along M, 32 rows each
    const int waveN = wave >> 2;  // 2 waves along N, 64 cols each

    const int blockN = blockIdx.x * BN;
    const int blockM = blockIdx.y * BM;

    // cooperative global->LDS mapping: 2 threads per row, 16 floats each
    const int grow  = tid >> 1;          // 0..127
    const int ghalf = (tid & 1) << 4;    // k offset 0 or 16

    const float* gA = data   + (size_t)(blockM + grow) * K + ghalf;
    const float* gW = weight + (size_t)(blockN + grow) * K + ghalf;
    const float* gM = wmask  + (size_t)(blockN + grow) * K + ghalf;

    const int wr = grow * LDSS + ghalf;  // LDS write offset (elements)

    const int frow = lane & 15;
    const int aOff = (waveM * 32 + frow) * LDSS + ((lane >> 4) << 4);
    const int bOff = (waveN * 64 + frow) * LDSS + ((lane >> 4) << 4);

    const v8f zacc = {0.f, 0.f, 0.f, 0.f, 0.f, 0.f, 0.f, 0.f};
    v8f acc[2][4];
#pragma unroll
    for (int mt = 0; mt < 2; ++mt)
#pragma unroll
        for (int nt = 0; nt < 4; ++nt) acc[mt][nt] = zacc;

    // ---- prologue: tile 0 ----
    float4 pa[4], pw[4], pm[4];
#pragma unroll
    for (int i = 0; i < 4; ++i) {
        pa[i] = *reinterpret_cast<const float4*>(gA + i * 4);
        pw[i] = *reinterpret_cast<const float4*>(gW + i * 4);
        pm[i] = *reinterpret_cast<const float4*>(gM + i * 4);
    }
    {
        float4 w0 = mul4(pw[0], pm[0]), w1 = mul4(pw[1], pm[1]);
        float4 w2 = mul4(pw[2], pm[2]), w3 = mul4(pw[3], pm[3]);
        *reinterpret_cast<v8bf*>(&As[0][wr])     = pack8(pa[0], pa[1]);
        *reinterpret_cast<v8bf*>(&As[0][wr + 8]) = pack8(pa[2], pa[3]);
        *reinterpret_cast<v8bf*>(&Bs[0][wr])     = pack8(w0, w1);
        *reinterpret_cast<v8bf*>(&Bs[0][wr + 8]) = pack8(w2, w3);
    }
    __syncthreads();

    const int NK = K / BK;  // 128 iterations
    for (int kt = 0; kt < NK; ++kt) {
        const int  cur = kt & 1;
        const bool pre = (kt + 1) < NK;

        if (pre) {  // register prefetch of next K-tile
            const int koff = (kt + 1) * BK;
#pragma unroll
            for (int i = 0; i < 4; ++i) {
                pa[i] = *reinterpret_cast<const float4*>(gA + koff + i * 4);
                pw[i] = *reinterpret_cast<const float4*>(gW + koff + i * 4);
                pm[i] = *reinterpret_cast<const float4*>(gM + koff + i * 4);
            }
        }

        v16bf af[2], bfr[4];
#pragma unroll
        for (int mt = 0; mt < 2; ++mt)
            af[mt] = load_fragA(&As[cur][aOff + mt * 16 * LDSS]);
#pragma unroll
        for (int nt = 0; nt < 4; ++nt)
            bfr[nt] = load_fragB(&Bs[cur][bOff + nt * 16 * LDSS]);

#pragma unroll
        for (int mt = 0; mt < 2; ++mt)
#pragma unroll
            for (int nt = 0; nt < 4; ++nt)
                acc[mt][nt] = __builtin_amdgcn_wmma_f32_16x16x32_bf16(
                    false, af[mt], false, bfr[nt], (short)0, acc[mt][nt],
                    false, false);

        if (pre) {  // fused mask, convert, stage into the other buffer
            const int nxt = cur ^ 1;
            float4 w0 = mul4(pw[0], pm[0]), w1 = mul4(pw[1], pm[1]);
            float4 w2 = mul4(pw[2], pm[2]), w3 = mul4(pw[3], pm[3]);
            *reinterpret_cast<v8bf*>(&As[nxt][wr])     = pack8(pa[0], pa[1]);
            *reinterpret_cast<v8bf*>(&As[nxt][wr + 8]) = pack8(pa[2], pa[3]);
            *reinterpret_cast<v8bf*>(&Bs[nxt][wr])     = pack8(w0, w1);
            *reinterpret_cast<v8bf*>(&Bs[nxt][wr + 8]) = pack8(w2, w3);
        }
        __syncthreads();
    }

    // ---- epilogue: bias add + store (f32 C layout: row = i + 8*(lane>=16), col = lane&15)
    const int col  = lane & 15;
    const int rsel = (lane >> 4) << 3;
#pragma unroll
    for (int nt = 0; nt < 4; ++nt) {
        const int   n = blockN + waveN * 64 + nt * 16 + col;
        const float b = bias[n];
#pragma unroll
        for (int mt = 0; mt < 2; ++mt) {
            const int m0 = blockM + waveM * 32 + mt * 16 + rsel;
#pragma unroll
            for (int i = 0; i < 8; ++i)
                out[(size_t)(m0 + i) * N + n] = acc[mt][nt][i] + b;
        }
    }
}

extern "C" void kernel_launch(void* const* d_in, const int* in_sizes, int n_in,
                              void* d_out, int out_size, void* d_ws, size_t ws_size,
                              hipStream_t stream) {
    const float* data   = (const float*)d_in[0];
    const float* weight = (const float*)d_in[1];
    const float* wmask  = (const float*)d_in[2];
    const float* bias   = (const float*)d_in[3];
    float* out = (float*)d_out;

    const int M = 4 * 2048;  // B*S
    const int N = 4096;      // D_OUT
    const int K = 4096;      // D_IN

    dim3 grid(N / BN, M / BM);  // 32 x 64 workgroups
    dim3 block(NTHREADS);
    hipLaunchKernelGGL(masked_linear_wmma_bf16, grid, block, 0, stream,
                       data, weight, wmask, bias, out, M, N, K);
}
